// SASRec_41111426957528
// MI455X (gfx1250) — compile-verified, matrix operand order checked by reading, not run on hardware
//
#include <hip/hip_runtime.h>
#include <hip/hip_bf16.h>
#include <hip/hip_fp16.h>

// ---------------------------------------------------------------------------
// SASRec forward for gfx1250 (MI455X). f16 WMMA (16x16x32) with f32 accum.
// GEMM K-loop uses ping-pong register buffers (fixed roles, no rotation
// copies): slot-A refill loads overlap slot-B WMMAs and vice versa.
// ---------------------------------------------------------------------------

typedef __attribute__((ext_vector_type(16))) _Float16 f16x16;
typedef __attribute__((ext_vector_type(8)))  _Float16 f16x8;
typedef __attribute__((ext_vector_type(8)))  float    f32x8;

#define BN 256      // batch
#define SN 256      // seq len
#define DN 256      // model dim
#define HN 4        // heads
#define DH 64       // head dim
#define MROWS (BN*SN)   // 65536 token rows

// ---- fragment helpers ------------------------------------------------------
// 16-bit A/B fragment layout (CDNA5 ISA 7.12.2): lane = {half = lane>>4, il = lane&15}
//   elements 0..7  : K = half*8 + 0..7
//   elements 8..15 : K = 16 + half*8 + 0..7
// A: il indexes M row; B (from transposed weight [N,K]): il indexes N col.
__device__ inline f16x16 load_frag(const _Float16* __restrict__ base, int ld,
                                   int row0, int col0, int lane) {
  const int hl = lane >> 4;
  const int il = lane & 15;
  const _Float16* p = base + (size_t)(row0 + il) * (size_t)ld + col0 + hl * 8;
  f16x8 lo = *(const f16x8*)p;
  f16x8 hi = *(const f16x8*)(p + 16);
  return __builtin_shufflevector(lo, hi, 0,1,2,3,4,5,6,7,8,9,10,11,12,13,14,15);
}

__device__ inline f32x8 wmma_f16(f16x16 a, f16x16 b, f32x8 c) {
  return __builtin_amdgcn_wmma_f32_16x16x32_f16(false, a, false, b, (short)0, c,
                                                false, false);
}

__device__ inline float gelu_exact(float x) {
  return 0.5f * x * (1.0f + erff(x * 0.70710678118654752f));
}

__device__ inline float wave_sum32(float v) {
  for (int o = 16; o >= 1; o >>= 1) v += __shfl_xor(v, o, 32);
  return v;
}

// ---- weight convert: f32 [K,N] -> f16 transposed [N,K] ---------------------
__global__ __launch_bounds__(256) void convw_k(const float* __restrict__ in,
                                               _Float16* __restrict__ out,
                                               int K, int N) {
  size_t i = (size_t)blockIdx.x * 256 + threadIdx.x;
  if (i < (size_t)K * N) {
    int kk = (int)(i / N);
    int nn = (int)(i % N);
    out[(size_t)nn * K + kk] = (_Float16)in[i];
  }
}

// ---- embedding + LayerNorm: one wave per token row -------------------------
__global__ __launch_bounds__(256) void embed_ln_k(const int* __restrict__ ids,
                                                  const float* __restrict__ item_emb,
                                                  const float* __restrict__ pos_emb,
                                                  _Float16* __restrict__ x16,
                                                  float* __restrict__ total) {
  const int wave = threadIdx.x >> 5;
  const int lane = threadIdx.x & 31;
  const size_t row = (size_t)blockIdx.x * 8 + wave;
  const int s = (int)(row & (SN - 1));
  const int id = ids[row];

  float e[8];
  const float* ip = item_emb + (size_t)id * DN + lane * 8;
  const float* pp = pos_emb + (size_t)s * DN + lane * 8;
  float lsum = 0.f;
#pragma unroll
  for (int i = 0; i < 8; ++i) { e[i] = ip[i] + pp[i]; lsum += e[i]; }
  const float mu = wave_sum32(lsum) * (1.0f / DN);
  float lvar = 0.f;
#pragma unroll
  for (int i = 0; i < 8; ++i) { float d = e[i] - mu; lvar += d * d; }
  const float rstd = rsqrtf(wave_sum32(lvar) * (1.0f / DN) + 1e-5f);

  _Float16* xo = x16 + row * DN + lane * 8;
  float* to = total + row * DN + lane * 8;
#pragma unroll
  for (int i = 0; i < 8; ++i) {
    float y = (e[i] - mu) * rstd;
    xo[i] = (_Float16)y;
    to[i] = y;              // total starts as x
  }
}

// ---- LayerNorm over f32 rows -> f16 (optionally accumulate into total) -----
__global__ __launch_bounds__(256) void ln_k(const float* __restrict__ in,
                                            _Float16* __restrict__ out16,
                                            float* __restrict__ total,
                                            int accum) {
  const int wave = threadIdx.x >> 5;
  const int lane = threadIdx.x & 31;
  const size_t row = (size_t)blockIdx.x * 8 + wave;
  const float* ip = in + row * DN + lane * 8;
  float e[8];
  float lsum = 0.f;
#pragma unroll
  for (int i = 0; i < 8; ++i) { e[i] = ip[i]; lsum += e[i]; }
  const float mu = wave_sum32(lsum) * (1.0f / DN);
  float lvar = 0.f;
#pragma unroll
  for (int i = 0; i < 8; ++i) { float d = e[i] - mu; lvar += d * d; }
  const float rstd = rsqrtf(wave_sum32(lvar) * (1.0f / DN) + 1e-5f);

  _Float16* xo = out16 + row * DN + lane * 8;
#pragma unroll
  for (int i = 0; i < 8; ++i) {
    float y = (e[i] - mu) * rstd;
    xo[i] = (_Float16)y;
    if (accum) total[row * DN + lane * 8 + i] += y;
  }
}

// ---- generic WMMA GEMM: out = epilogue(A[M,K] @ Wt[N,K]^T + bias) ----------
// 8 waves/block; wave (w>>2) picks M sub-tile, (w&3) picks N strip of 64.
// Block covers 32 rows x N cols. grid.x = MROWS/32. Requires K % 64 == 0.
// Ping-pong fragment buffers: consume slot A (k), refill slot A for k+64,
// consume slot B (k+32), refill slot B for k+96. Tail prefetch indices clamp
// to 0/32 (harmless redundant loads) to keep the loop branchless.
#define EPI_F16     0
#define EPI_GELU16  1
#define EPI_RES32   2
#define EPI_F32     3

__global__ __launch_bounds__(256) void gemm_ep(const _Float16* __restrict__ A,
                                               const _Float16* __restrict__ Wt,
                                               const float* __restrict__ bias,
                                               const _Float16* __restrict__ res,
                                               _Float16* __restrict__ out16,
                                               float* __restrict__ out32,
                                               int K, int N, int mode) {
  const int wave = threadIdx.x >> 5;
  const int lane = threadIdx.x & 31;
  const int hl = lane >> 4;
  const int nl = lane & 15;
  const int row0 = (blockIdx.x * 2 + (wave >> 2)) * 16;

  for (int n0 = (wave & 3) * 64; n0 < N; n0 += 256) {
    f32x8 acc[4];
#pragma unroll
    for (int j = 0; j < 4; ++j) acc[j] = (f32x8){0,0,0,0,0,0,0,0};

    // prologue: slot A holds k=0, slot B holds k=32
    f16x16 aA = load_frag(A, K, row0, 0, lane);
    f16x16 aB = load_frag(A, K, row0, 32, lane);
    f16x16 bA[4], bB[4];
#pragma unroll
    for (int j = 0; j < 4; ++j) {
      bA[j] = load_frag(Wt, K, n0 + j * 16, 0, lane);
      bB[j] = load_frag(Wt, K, n0 + j * 16, 32, lane);
    }

    for (int k0 = 0; k0 < K; k0 += 64) {
      int kA = k0 + 64; kA = (kA < K) ? kA : 0;   // next fill for slot A
      int kB = k0 + 96; kB = (kB < K) ? kB : 32;  // next fill for slot B

      // consume slot A, then refill it (loads overlap slot-B WMMAs)
#pragma unroll
      for (int j = 0; j < 4; ++j) acc[j] = wmma_f16(aA, bA[j], acc[j]);
      aA = load_frag(A, K, row0, kA, lane);
#pragma unroll
      for (int j = 0; j < 4; ++j) bA[j] = load_frag(Wt, K, n0 + j * 16, kA, lane);

      // consume slot B, then refill it (loads overlap next slot-A WMMAs)
#pragma unroll
      for (int j = 0; j < 4; ++j) acc[j] = wmma_f16(aB, bB[j], acc[j]);
      aB = load_frag(A, K, row0, kB, lane);
#pragma unroll
      for (int j = 0; j < 4; ++j) bB[j] = load_frag(Wt, K, n0 + j * 16, kB, lane);
    }

#pragma unroll
    for (int j = 0; j < 4; ++j) {
      const int col = n0 + j * 16 + nl;
      const float bv = bias[col];
#pragma unroll
      for (int r = 0; r < 8; ++r) {
        const size_t row = (size_t)row0 + r + 8 * hl;
        float v = acc[j][r] + bv;
        if (mode == EPI_GELU16) v = gelu_exact(v);
        if (mode == EPI_RES32)  v += (float)res[row * N + col];
        if (mode <= EPI_GELU16) out16[row * (size_t)N + col] = (_Float16)v;
        else                    out32[row * (size_t)N + col] = v;
      }
    }
  }
}

// ---- fused attention: one block per (b,h); K,V^T staged in LDS -------------
__global__ __launch_bounds__(256) void attn_k(const _Float16* __restrict__ q,
                                              const _Float16* __restrict__ k,
                                              const _Float16* __restrict__ v,
                                              const int* __restrict__ ids,
                                              _Float16* __restrict__ ctx) {
  __shared__ __align__(16) _Float16 Klds[SN * DH];      // 32 KB, layout [s][dh]
  __shared__ __align__(16) _Float16 Vtlds[DH * SN];     // 32 KB, layout [dh][s]
  __shared__ __align__(16) _Float16 Probs[8 * 16 * SN]; // 64 KB, per-wave 16xS
  __shared__ float padb[SN];                            // key pad bias

  const int tid = threadIdx.x;
  const int wave = tid >> 5;
  const int lane = tid & 31;
  const int hl = lane >> 4;
  const int nl = lane & 15;
  const int b = blockIdx.x >> 2;
  const int h = blockIdx.x & 3;

  const _Float16* kbase = k + ((size_t)b * SN) * DN + h * DH;
  const _Float16* vbase = v + ((size_t)b * SN) * DN + h * DH;
  const _Float16* qbase = q + ((size_t)b * SN) * DN + h * DH;
  _Float16* cbase = ctx + ((size_t)b * SN) * DN + h * DH;

  // stage K (natural) into LDS, 8 f16 per load
  for (int e = tid; e < SN * DH / 8; e += 256) {
    int s_ = e >> 3, d8 = (e & 7) * 8;
    *(f16x8*)(&Klds[s_ * DH + d8]) = *(const f16x8*)(kbase + (size_t)s_ * DN + d8);
  }
  // stage V transposed into LDS
  for (int e = tid; e < SN * DH; e += 256) {
    int s_ = e >> 6, d = e & 63;
    Vtlds[d * SN + s_] = vbase[(size_t)s_ * DN + d];
  }
  if (tid < SN) padb[tid] = (ids[(size_t)b * SN + tid] > 0) ? 0.f : -10000.0f;
  __syncthreads();

  _Float16* pw = &Probs[wave * 16 * SN];
  const float scale = 0.125f; // 1/sqrt(64)

  for (int it = 0; it < 2; ++it) {
    const int q0 = (wave + it * 8) * 16;

    // scores: Q[16,64] x K^T[64,256] -> 16 accumulators
    f32x8 sc[16];
#pragma unroll
    for (int t = 0; t < 16; ++t) sc[t] = (f32x8){0,0,0,0,0,0,0,0};
#pragma unroll
    for (int kk = 0; kk < 2; ++kk) {
      f16x16 aq = load_frag(qbase, DN, q0, kk * 32, lane);
#pragma unroll
      for (int t = 0; t < 16; ++t) {
        f16x16 bk = load_frag(Klds, DH, t * 16, kk * 32, lane);
        sc[t] = wmma_f16(aq, bk, sc[t]);
      }
    }

    // mask + softmax; row m = r + 8*hl is held by the 16 lanes of this half
#pragma unroll
    for (int r = 0; r < 8; ++r) {
      const int qi = q0 + r + 8 * hl;
      float mx = -3.4e38f;
#pragma unroll
      for (int t = 0; t < 16; ++t) {
        const int j = t * 16 + nl;
        float sv = sc[t][r] * scale + ((j <= qi) ? padb[j] : -10000.0f);
        sc[t][r] = sv;
        mx = fmaxf(mx, sv);
      }
      for (int o = 1; o < 16; o <<= 1) mx = fmaxf(mx, __shfl_xor(mx, o, 16));
      float sum = 0.f;
#pragma unroll
      for (int t = 0; t < 16; ++t) {
        float p = __expf(sc[t][r] - mx);
        sc[t][r] = p;
        sum += p;
      }
      for (int o = 1; o < 16; o <<= 1) sum += __shfl_xor(sum, o, 16);
      const float inv = 1.0f / sum;
#pragma unroll
      for (int t = 0; t < 16; ++t)
        pw[(r + 8 * hl) * SN + t * 16 + nl] = (_Float16)(sc[t][r] * inv);
    }

    // ctx: P[16,256] x V[256,64] (B from V^T in LDS)
    f32x8 o4[4];
#pragma unroll
    for (int j = 0; j < 4; ++j) o4[j] = (f32x8){0,0,0,0,0,0,0,0};
    for (int k0 = 0; k0 < SN; k0 += 32) {
      f16x16 ap = load_frag(pw, SN, 0, k0, lane);
#pragma unroll
      for (int j = 0; j < 4; ++j) {
        f16x16 bv = load_frag(Vtlds, SN, j * 16, k0, lane);
        o4[j] = wmma_f16(ap, bv, o4[j]);
      }
    }
#pragma unroll
    for (int j = 0; j < 4; ++j) {
      const int col = j * 16 + nl;
#pragma unroll
      for (int r = 0; r < 8; ++r) {
        const size_t row = (size_t)q0 + r + 8 * hl;
        cbase[row * DN + col] = (_Float16)o4[j][r];
      }
    }
  }
}

// ---------------------------------------------------------------------------
extern "C" void kernel_launch(void* const* d_in, const int* in_sizes, int n_in,
                              void* d_out, int out_size, void* d_ws, size_t ws_size,
                              hipStream_t stream) {
  (void)in_sizes; (void)n_in; (void)out_size; (void)ws_size;

  const int*   ids      = (const int*)d_in[0];
  const float* item_emb = (const float*)d_in[1];
  const float* pos_emb  = (const float*)d_in[2];
  const float* Wq = (const float*)d_in[3];
  const float* bq = (const float*)d_in[4];
  const float* Wk = (const float*)d_in[5];
  const float* bk = (const float*)d_in[6];
  const float* Wv = (const float*)d_in[7];
  const float* bv = (const float*)d_in[8];
  const float* Wo = (const float*)d_in[9];
  const float* bo = (const float*)d_in[10];
  const float* W1 = (const float*)d_in[11];
  const float* b1 = (const float*)d_in[12];
  const float* W2 = (const float*)d_in[13];
  const float* b2 = (const float*)d_in[14];
  float* out = (float*)d_out;

  char* ws = (char*)d_ws;
  const size_t MBy = 1ull << 20;
  _Float16* X16  = (_Float16*)(ws + 0);          // 32 MB
  _Float16* Q16  = (_Float16*)(ws + 32 * MBy);   // 32 MB (part of FFN region)
  _Float16* K16  = (_Float16*)(ws + 64 * MBy);   // 32 MB (part of FFN region)
  _Float16* V16  = (_Float16*)(ws + 96 * MBy);   // 32 MB (part of FFN region)
  _Float16* C16  = (_Float16*)(ws + 128 * MBy);  // 32 MB (part of FFN region)
  _Float16* FFN  = Q16;                          // 128 MB reuse of Q/K/V/C
  _Float16* H16  = (_Float16*)(ws + 160 * MBy);  // 32 MB
  float*    T32  = (float*)  (ws + 192 * MBy);   // 64 MB
  _Float16* WT   = (_Float16*)(ws + 256 * MBy);  // ~3 MB f16 weights

  const size_t perL = 4 * 65536 + 2 * 262144;    // f16 elems per layer
  const int DD = DN * DN;            // 65536
  const int DF = DN * 4 * DN;        // 262144

  // convert + transpose all weights to f16 [N,K]
  for (int l = 0; l < 2; ++l) {
    _Float16* wqt = WT + l * perL;
    _Float16* wkt = wqt + 65536;
    _Float16* wvt = wkt + 65536;
    _Float16* wot = wvt + 65536;
    _Float16* w1t = wot + 65536;
    _Float16* w2t = w1t + 262144;
    convw_k<<<(DD + 255) / 256, 256, 0, stream>>>(Wq + (size_t)l * DD, wqt, DN, DN);
    convw_k<<<(DD + 255) / 256, 256, 0, stream>>>(Wk + (size_t)l * DD, wkt, DN, DN);
    convw_k<<<(DD + 255) / 256, 256, 0, stream>>>(Wv + (size_t)l * DD, wvt, DN, DN);
    convw_k<<<(DD + 255) / 256, 256, 0, stream>>>(Wo + (size_t)l * DD, wot, DN, DN);
    convw_k<<<(DF + 255) / 256, 256, 0, stream>>>(W1 + (size_t)l * DF, w1t, DN, 4 * DN);
    convw_k<<<(DF + 255) / 256, 256, 0, stream>>>(W2 + (size_t)l * DF, w2t, 4 * DN, DN);
  }

  // x = LN(item_emb[ids] + pos); total = x
  embed_ln_k<<<MROWS / 8, 256, 0, stream>>>(ids, item_emb, pos_emb, X16, out);

  for (int l = 0; l < 2; ++l) {
    _Float16* wqt = WT + l * perL;
    _Float16* wkt = wqt + 65536;
    _Float16* wvt = wkt + 65536;
    _Float16* wot = wvt + 65536;
    _Float16* w1t = wot + 65536;
    _Float16* w2t = w1t + 262144;

    // QKV projections
    gemm_ep<<<MROWS / 32, 256, 0, stream>>>(X16, wqt, bq + l * DN, nullptr,
                                            Q16, nullptr, DN, DN, EPI_F16);
    gemm_ep<<<MROWS / 32, 256, 0, stream>>>(X16, wkt, bk + l * DN, nullptr,
                                            K16, nullptr, DN, DN, EPI_F16);
    gemm_ep<<<MROWS / 32, 256, 0, stream>>>(X16, wvt, bv + l * DN, nullptr,
                                            V16, nullptr, DN, DN, EPI_F16);
    // fused masked attention
    attn_k<<<BN * HN, 256, 0, stream>>>(Q16, K16, V16, ids, C16);
    // O projection + residual -> f32, then LN -> h
    gemm_ep<<<MROWS / 32, 256, 0, stream>>>(C16, wot, bo + l * DN, X16,
                                            nullptr, T32, DN, DN, EPI_RES32);
    ln_k<<<MROWS / 8, 256, 0, stream>>>(T32, H16, nullptr, 0);
    // FFN
    gemm_ep<<<MROWS / 32, 256, 0, stream>>>(H16, w1t, b1 + l * 4 * DN, nullptr,
                                            FFN, nullptr, DN, 4 * DN, EPI_GELU16);
    gemm_ep<<<MROWS / 32, 256, 0, stream>>>(FFN, w2t, b2 + l * DN, nullptr,
                                            nullptr, T32, 4 * DN, DN, EPI_F32);
    // x = LN(ffn); total += x
    ln_k<<<MROWS / 8, 256, 0, stream>>>(T32, X16, out, 1);
  }
}
